// countAndRecLayer_77025943486831
// MI455X (gfx1250) — compile-verified, matrix operand order checked by reading, not run on hardware
//
#include <hip/hip_runtime.h>

#define BATCH 16
#define HH    512
#define WW    512
#define HW    (HH * WW)
#define NTOT  (BATCH * HW)

#define TS    32      // tile size for reconstruction sweeps
#define INNER 40      // Jacobi relaxations per tile per launch

#define NEG_BIG (-3.402823466e38f)

typedef __attribute__((ext_vector_type(16))) _Float16 v16h;
typedef __attribute__((ext_vector_type(8)))  float    v8f;

// ---- CDNA5 async global->LDS path (guarded; falls back to scalar loads) ----
#if defined(__has_builtin)
#if __has_builtin(__builtin_amdgcn_global_load_async_to_lds_b32)
#define HAVE_ASYNC_LDS 1
#endif
#endif
#ifndef HAVE_ASYNC_LDS
#define HAVE_ASYNC_LDS 0
#endif

#if HAVE_ASYNC_LDS
typedef __attribute__((address_space(1))) int gbl_int_t;   // global (AS1)
typedef __attribute__((address_space(3))) int lds_int_t;   // LDS    (AS3)

__device__ __forceinline__ void async_g2l_b32(const float* g, float* l) {
    int* gi = (int*)(void*)const_cast<float*>(g);
    int* li = (int*)(void*)l;
    __builtin_amdgcn_global_load_async_to_lds_b32(
        (gbl_int_t*)gi, (lds_int_t*)li, /*imm_offset=*/0, /*cpol=*/0);
}
__device__ __forceinline__ void wait_async0() {
#if __has_builtin(__builtin_amdgcn_s_wait_asynccnt)
    __builtin_amdgcn_s_wait_asynccnt(0);
#else
    asm volatile("s_wait_asynccnt 0" ::: "memory");
#endif
}
#endif

// ---------------------------------------------------------------- utilities
__global__ void k_zero_small(float* cc, int* flags) {
    int t = threadIdx.x;
    if (t < 16) cc[t] = 0.0f;
    if (t == 16) { flags[0] = 0; flags[1] = 0; }
}

// cur = x - h[b]  (marker for reconstruction #1)
__global__ void k_init_marker(const float* __restrict__ x,
                              const float* __restrict__ h,
                              float* __restrict__ cur) {
    int i = blockIdx.x * blockDim.x + threadIdx.x;
    if (i < NTOT) {
        int b = i / HW;           // C == 1
        cur[i] = x[i] - h[b];
    }
}

// One outer sweep of reconstruction-by-dilation: monotone in-place chaotic
// relaxation. Each block: load 34x34 (tile+halo) into LDS (async DMA on the
// interior fast path), run INNER local Jacobi sweeps of
// m <- min(max3x3(m), mask), write tile back.
__global__ __launch_bounds__(256)
void k_recon_sweep(float* __restrict__ cur, const float* __restrict__ mask) {
    __shared__ float s[(TS + 2) * (TS + 2)];
    const int  b   = blockIdx.z;
    const int  tx0 = blockIdx.x * TS;
    const int  ty0 = blockIdx.y * TS;
    const int  lx  = threadIdx.x;        // 0..31
    const int  lyb = threadIdx.y;        // 0..7
    const int  tid = lyb * 32 + lx;
    const long base = (long)b * HW;

    // CDNA5 prefetch path (global_prefetch_b8) for the mask tile
    __builtin_prefetch(&mask[base + (long)(ty0 + lyb) * WW + tx0 + lx], 0, 3);

    const bool interior =
        (tx0 > 0) && (ty0 > 0) && (tx0 + TS < WW) && (ty0 + TS < HH);

#if HAVE_ASYNC_LDS
    if (interior) {
        // fast path: whole halo tile in-bounds -> async DMA straight to LDS
        for (int i = tid; i < (TS + 2) * (TS + 2); i += 256) {
            int ly  = i / (TS + 2);
            int lxx = i % (TS + 2);
            async_g2l_b32(&cur[base + (long)(ty0 + ly - 1) * WW + (tx0 + lxx - 1)],
                          &s[i]);
        }
    } else
#endif
    {
        // boundary path: scalar bounds-checked loads; outside image = -inf
        for (int i = tid; i < (TS + 2) * (TS + 2); i += 256) {
            int ly  = i / (TS + 2);
            int lxx = i % (TS + 2);
            int gy  = ty0 + ly - 1;
            int gx  = tx0 + lxx - 1;
            float v = NEG_BIG;
            if (gy >= 0 && gy < HH && gx >= 0 && gx < WW)
                v = cur[base + (long)gy * WW + gx];
            s[i] = v;
        }
    }

    // per-thread mask values (4 rows, stride 8)
    float mk[4];
#pragma unroll
    for (int j = 0; j < 4; ++j)
        mk[j] = mask[base + (long)(ty0 + lyb + 8 * j) * WW + tx0 + lx];

#if HAVE_ASYNC_LDS
    if (interior) wait_async0();   // per-wave: async LDS writes complete
#endif
    __syncthreads();               // cross-wave visibility of the LDS tile

    for (int it = 0; it < INNER; ++it) {
        float nv[4];
#pragma unroll
        for (int j = 0; j < 4; ++j) {
            int y  = lyb + 8 * j + 1;
            int xx = lx + 1;
            const float* r0 = &s[(y - 1) * (TS + 2) + xx - 1];
            const float* r1 = &s[(y    ) * (TS + 2) + xx - 1];
            const float* r2 = &s[(y + 1) * (TS + 2) + xx - 1];
            float m = fmaxf(fmaxf(fmaxf(r0[0], r0[1]), fmaxf(r0[2], r1[0])),
                            fmaxf(fmaxf(r1[1], r1[2]),
                                  fmaxf(fmaxf(r2[0], r2[1]), r2[2])));
            nv[j] = fminf(m, mk[j]);   // >= old value: monotone
        }
        __syncthreads();
#pragma unroll
        for (int j = 0; j < 4; ++j)
            s[(lyb + 8 * j + 1) * (TS + 2) + lx + 1] = nv[j];
        __syncthreads();
    }

#pragma unroll
    for (int j = 0; j < 4; ++j)
        cur[base + (long)(ty0 + lyb + 8 * j) * WW + tx0 + lx] =
            s[(lyb + 8 * j + 1) * (TS + 2) + lx + 1];
}

// xh = cur ; cur = cur - eps  (marker for reconstruction #2, mask = xh)
__global__ void k_finish_recon1(float* __restrict__ cur, float* __restrict__ xh) {
    int i = blockIdx.x * blockDim.x + threadIdx.x;
    if (i < NTOT) {
        float v = cur[i];
        xh[i]  = v;
        cur[i] = v - 1e-5f;
    }
}

// Rmax = (xh - rec2 > 0); global any0/any1 flags; cur = min(u, Rmax)
__global__ void k_rmax_and_marker3(const float* __restrict__ xh,
                                   float* __restrict__ cur,
                                   const float* __restrict__ u,
                                   float* __restrict__ rmax,
                                   int* __restrict__ flags) {
    __shared__ int sAny1, sAny0;
    int i = blockIdx.x * blockDim.x + threadIdx.x;
    if (threadIdx.x == 0) { sAny1 = 0; sAny0 = 0; }
    __syncthreads();
    if (i < NTOT) {
        float rec2 = cur[i];
        float r = (xh[i] - rec2 > 0.0f) ? 1.0f : 0.0f;
        rmax[i] = r;
        if (r > 0.5f) sAny1 = 1; else sAny0 = 1;   // benign race
        cur[i] = fminf(u[i], r);
    }
    __syncthreads();
    if (threadIdx.x == 0) {
        if (sAny1) atomicOr(&flags[0], 1);
        if (sAny0) atomicOr(&flags[1], 1);
    }
}

// CC[b] = sum(u == R) via WMMA: A = 512 detection bits/wave (f16, exact),
// B = ones; with B==1, sum(C) = 16 * sum(A), layout-independent.
__global__ __launch_bounds__(256)
void k_detect_reduce(const float* __restrict__ u,
                     const float* __restrict__ R,
                     float* __restrict__ cc) {
    const int  b    = blockIdx.y;
    const long base = (long)b * HW;
    const int  lane = threadIdx.x & 31;
    const int  wid  = blockIdx.x * 8 + (threadIdx.x >> 5);   // 0..127 per batch

    v16h ones;
#pragma unroll
    for (int k = 0; k < 16; ++k) ones[k] = (_Float16)1.0f;

    v8f c = {};
#pragma unroll
    for (int ch = 0; ch < 4; ++ch) {          // 128 waves * 4 * 512 = 262144
        long off = base + ((long)wid * 4 + ch) * 512 + (long)lane * 16;
        v16h a;
#pragma unroll
        for (int k = 0; k < 16; ++k) {
            float uv = u[off + k];
            float rv = R[off + k];
            a[k] = (uv == rv) ? (_Float16)1.0f : (_Float16)0.0f;
        }
        c = __builtin_amdgcn_wmma_f32_16x16x32_f16(
                /*neg_a=*/false, a, /*neg_b=*/false, ones,
                /*c_mod=*/(short)0, c, /*reuse_a=*/false, /*reuse_b=*/false);
    }
    float p = c[0] + c[1] + c[2] + c[3] + c[4] + c[5] + c[6] + c[7];
#pragma unroll
    for (int o = 16; o >= 1; o >>= 1)
        p += __shfl_xor(p, o, 32);
    if (lane == 0)
        atomicAdd(&cc[b], p * 0.0625f);       // /16: row-sum replication
}

// CC_ = min(CC, 100*(max(Rmax)-min(Rmax))*CC)
__global__ void k_final(const float* __restrict__ cc,
                        const int* __restrict__ flags,
                        float* __restrict__ out) {
    int i = threadIdx.x;
    if (i < 16) {
        float gmax = flags[0] ? 1.0f : 0.0f;   // any pixel == 1
        float gmin = flags[1] ? 0.0f : 1.0f;   // any pixel == 0
        float v = cc[i];
        out[i] = fminf(v, 100.0f * (gmax - gmin) * v);
    }
}

extern "C" void kernel_launch(void* const* d_in, const int* in_sizes, int n_in,
                              void* d_out, int out_size, void* d_ws, size_t ws_size,
                              hipStream_t stream) {
    (void)in_sizes; (void)n_in; (void)out_size; (void)ws_size;

    const float* x = (const float*)d_in[0];
    const float* h = (const float*)d_in[1];
    const float* u = (const float*)d_in[2];
    float* out  = (float*)d_out;
    float* xh   = out + 16;                     // output xh region
    float* ws   = (float*)d_ws;
    float* cur  = ws;                           // 16 MB: marker buffer
    float* rmax = ws + (size_t)NTOT;            // 16 MB: Rmax / mask 3
    float* cc   = ws + 2 * (size_t)NTOT;        // 16 floats
    int*   flags = (int*)(cc + 16);             // 2 ints

    dim3 eb(256), eg((NTOT + 255) / 256);
    dim3 sb(32, 8), sg(WW / TS, HH / TS, BATCH);

    hipLaunchKernelGGL(k_zero_small, dim3(1), dim3(32), 0, stream, cc, flags);

    // reconstruction #1: xh = reconstruct(x - h, x)
    hipLaunchKernelGGL(k_init_marker, eg, eb, 0, stream, x, h, cur);
    for (int i = 0; i < 24; ++i)
        hipLaunchKernelGGL(k_recon_sweep, sg, sb, 0, stream, cur, x);
    hipLaunchKernelGGL(k_finish_recon1, eg, eb, 0, stream, cur, xh);

    // reconstruction #2: reconstruct(xh - eps, xh)
    for (int i = 0; i < 12; ++i)
        hipLaunchKernelGGL(k_recon_sweep, sg, sb, 0, stream, cur, (const float*)xh);

    // Rmax, flags, marker for reconstruction #3
    hipLaunchKernelGGL(k_rmax_and_marker3, eg, eb, 0, stream,
                       (const float*)xh, cur, u, rmax, flags);

    // reconstruction #3: R = reconstruct(min(u,Rmax), Rmax)
    for (int i = 0; i < 12; ++i)
        hipLaunchKernelGGL(k_recon_sweep, sg, sb, 0, stream, cur, (const float*)rmax);

    // Detection count via WMMA + final CC_
    hipLaunchKernelGGL(k_detect_reduce, dim3(16, BATCH), dim3(256), 0, stream,
                       u, (const float*)cur, cc);
    hipLaunchKernelGGL(k_final, dim3(1), dim3(16), 0, stream, cc, flags, out);
}
